// DecentralController_1425929142612
// MI455X (gfx1250) — compile-verified, matrix operand order checked by reading, not run on hardware
//
#include <hip/hip_runtime.h>

// ---------------------------------------------------------------------------
// DecentralController forward pass for MI455X (gfx1250, wave32, WMMA).
//
// Heavy stages (5 convs + compressor FC) run as implicit-GEMM on
// v_wmma_f32_16x16x32_f16 (f16 operands, f32 accumulate). BN/bias/LeakyReLU
// fused into WMMA epilogue. Compressor GEMM stages A-rows through LDS with
// GLOBAL_LOAD_ASYNC_TO_LDS_B128 (ASYNCcnt double-buffer) when the toolchain
// exposes the builtins. Tiny graph-filter / MLP stages run on VALU.
// ---------------------------------------------------------------------------

typedef _Float16 half_t;
typedef __attribute__((ext_vector_type(16))) _Float16 v16h;
typedef __attribute__((ext_vector_type(8)))  _Float16 v8h;
typedef __attribute__((ext_vector_type(8)))  float    v8f;
typedef int b128_t __attribute__((vector_size(16)));   // async-LDS builtin operand

#define LEAK 0.01f
#define MAX_KPAD 576    // max Cin*9 (layers 3/4: 64*9=576), multiple of 32
#define CONV_NPIX 32    // pixels per conv workgroup (2 WMMA N-subtiles)

#if __has_builtin(__builtin_amdgcn_global_load_async_to_lds_b128) && \
    __has_builtin(__builtin_amdgcn_s_wait_asynccnt)
#define HAVE_ASYNC_LDS 1
#else
#define HAVE_ASYNC_LDS 0
#endif

__device__ __forceinline__ float leakyf(float v) { return v > 0.f ? v : LEAK * v; }

__device__ __forceinline__ v16h cat8(v8h lo, v8h hi) {
  return __builtin_shufflevector(lo, hi, 0, 1, 2, 3, 4, 5, 6, 7,
                                 8, 9, 10, 11, 12, 13, 14, 15);
}

// ---------------------------------------------------------------- utilities
__global__ void cvt_f32_f16_kernel(const float* __restrict__ s,
                                   half_t* __restrict__ d, int n) {
  int i = blockIdx.x * blockDim.x + threadIdx.x;
  if (i < n) d[i] = (half_t)s[i];
}

// Fold conv bias + BatchNorm into per-channel scale/shift:
//   out = conv*scale + shift
__global__ void prep_bn_kernel(const float* __restrict__ bias,
                               const float* __restrict__ gamma,
                               const float* __restrict__ beta,
                               const float* __restrict__ mean,
                               const float* __restrict__ var,
                               float* __restrict__ scale,
                               float* __restrict__ shift, int C) {
  int c = blockIdx.x * blockDim.x + threadIdx.x;
  if (c < C) {
    float inv = gamma[c] * rsqrtf(var[c] + 1e-5f);
    scale[c] = inv;
    shift[c] = (bias[c] - mean[c]) * inv + beta[c];
  }
}

// ------------------------------------------------- weight fragment packing
// Per-lane WMMA A fragments: [mtile][kchunk][lane][16 halves] contiguous
// (32B per lane -> 2x global_load_b128). 16-bit A layout: lanes 0-15 hold
// M=lane, K={0..7,16..23}; lanes 16-31 hold M=lane-16, K={8..15,24..31}.
__global__ void repack_conv_w_kernel(const float* __restrict__ w,
                                     half_t* __restrict__ pack,
                                     int Cin, int Cout, int Kfull, int nKC) {
  int idx = blockIdx.x * blockDim.x + threadIdx.x;
  int mtiles = Cout >> 4;
  int total = mtiles * nKC * 32;
  if (idx >= total) return;
  int lane = idx & 31;
  int kc   = (idx >> 5) % nKC;
  int mt   = (idx >> 5) / nKC;
  int row  = mt * 16 + (lane & 15);
  int koff = (lane < 16) ? 0 : 8;
  half_t* dst = pack + (size_t)idx * 16;
#pragma unroll
  for (int j = 0; j < 16; ++j) {
    int k = kc * 32 + koff + (j < 8 ? j : j + 8);
    float v = 0.f;
    if (k < Kfull && row < Cout) {
      int c = k / 9, r = k - c * 9;
      v = w[((size_t)row * Cin + c) * 9 + r];
    }
    dst[j] = (half_t)v;
  }
}

// Pack FC weight [K][N] (f32 row-major) into per-lane WMMA B fragments:
// [ntile][kchunk][lane][16 halves]; lane holds column nt*16+(lane&15).
__global__ void repack_gemm_b_kernel(const float* __restrict__ w,
                                     half_t* __restrict__ pack,
                                     int K, int N, int nKC) {
  int idx = blockIdx.x * blockDim.x + threadIdx.x;
  int ntiles = N >> 4;
  int total = ntiles * nKC * 32;
  if (idx >= total) return;
  int lane = idx & 31;
  int kc   = (idx >> 5) % nKC;
  int nt   = (idx >> 5) / nKC;
  int col  = nt * 16 + (lane & 15);
  int koff = (lane < 16) ? 0 : 8;
  half_t* dst = pack + (size_t)idx * 16;
#pragma unroll
  for (int j = 0; j < 16; ++j) {
    int k = kc * 32 + koff + (j < 8 ? j : j + 8);
    dst[j] = (half_t)((k < K && col < N) ? w[(size_t)k * N + col] : 0.f);
  }
}

// ------------------------------------------------------- implicit-GEMM conv
// Grid: (ceil(H*W/32), 320 images). Block: (32, Cout/16); one wave owns a
// 16(Cout)x32(pixel) output tile = 2 WMMA N-subtiles sharing one A fragment.
// Shared im2col strip Bt[32 px][Kpad] built once per block; the fill loop
// gives each thread one k (one (c,dy,dx)) and walks pixels incrementally so
// the div/mod address math is hoisted out of the inner loop.
__global__ __launch_bounds__(256) void conv_wmma_kernel(
    const half_t* __restrict__ in,      // [IMG][Cin][H][W]
    const half_t* __restrict__ wpack,   // packed A fragments
    const float* __restrict__ scale, const float* __restrict__ shift,
    half_t* __restrict__ out,           // [IMG][Cout][H][W]
    int Cin, int Cout, int H, int W, int Kfull, int nKC) {
  __shared__ __attribute__((aligned(32))) half_t Bt[CONV_NPIX * MAX_KPAD];
  const int Kpad  = nKC * 32;
  const int img   = blockIdx.y;
  const int tile0 = blockIdx.x * CONV_NPIX;
  const int lane  = threadIdx.x;
  const int wave  = threadIdx.y;
  const int nth   = blockDim.x * blockDim.y;
  const int tid   = wave * 32 + lane;
  const int HW    = H * W;
  const half_t* inimg = in + (size_t)img * Cin * HW;

  // Cooperative im2col into LDS (zero-padded SAME conv; K padded to 32).
  for (int kk = tid; kk < Kpad; kk += nth) {
    if (kk >= Kfull) {
      for (int p = 0; p < CONV_NPIX; ++p) Bt[p * Kpad + kk] = (half_t)0.f;
    } else {
      int c = kk / 9, r = kk - c * 9;
      int dy = r / 3 - 1, dx = (r % 3) - 1;
      const half_t* cin = inimg + (size_t)c * HW;
      int pixel = tile0;
      int y = pixel / W, x = pixel - y * W;
      for (int p = 0; p < CONV_NPIX; ++p) {
        half_t v = (half_t)0.f;
        if (pixel < HW) {
          int yy = y + dy, xx = x + dx;
          if (yy >= 0 && yy < H && xx >= 0 && xx < W) v = cin[yy * W + xx];
        }
        Bt[p * Kpad + kk] = v;
        ++pixel; ++x;
        if (x == W) { x = 0; ++y; }
      }
    }
  }
  __syncthreads();

  const int colr = lane & 15;
  const int koff = (lane < 16) ? 0 : 8;
  const half_t* apack = wpack + ((size_t)wave * nKC * 32 + lane) * 16;

  v8f acc0 = {}, acc1 = {};
  for (int kc = 0; kc < nKC; ++kc) {
    if (kc + 1 < nKC)  // gfx1250 global_prefetch_b8 for next A fragment
      __builtin_prefetch(apack + (size_t)(kc + 1) * 32 * 16, 0, 0);
    v16h a = *(const v16h*)(apack + (size_t)kc * 32 * 16);
    const half_t* bp0 = &Bt[colr * Kpad + kc * 32 + koff];
    const half_t* bp1 = &Bt[(16 + colr) * Kpad + kc * 32 + koff];
    v16h b0 = cat8(*(const v8h*)bp0, *(const v8h*)(bp0 + 16));
    v16h b1 = cat8(*(const v8h*)bp1, *(const v8h*)(bp1 + 16));
    acc0 = __builtin_amdgcn_wmma_f32_16x16x32_f16(
        false, a, false, b0, (short)0, acc0, false, false);
    acc1 = __builtin_amdgcn_wmma_f32_16x16x32_f16(
        false, a, false, b1, (short)0, acc1, false, false);
  }

  // Epilogue: fused BN + LeakyReLU. C/D layout: lane holds column N=lane&15;
  // VGPR r holds row M = r (+8 for lanes 16-31).
  int mbase = wave * 16 + ((lane < 16) ? 0 : 8);
  int pix0 = tile0 + colr;
  int pix1 = tile0 + 16 + colr;
#pragma unroll
  for (int r = 0; r < 8; ++r) {
    int cout = mbase + r;
    float sc = scale[cout], sh = shift[cout];
    size_t obase = ((size_t)img * Cout + cout) * HW;
    if (pix0 < HW) out[obase + pix0] = (half_t)leakyf(acc0[r] * sc + sh);
    if (pix1 < HW) out[obase + pix1] = (half_t)leakyf(acc1[r] * sc + sh);
  }
}

// ------------------------------------------------------------- 2x2 maxpool
__global__ void maxpool2_kernel(const half_t* __restrict__ in,
                                half_t* __restrict__ out,
                                int C, int H, int W) {
  int Ho = H >> 1, Wo = W >> 1;
  int img = blockIdx.y;
  int i = blockIdx.x * blockDim.x + threadIdx.x;
  int tot = C * Ho * Wo;
  if (i >= tot) return;
  int c = i / (Ho * Wo);
  int rem = i - c * Ho * Wo;
  int y = rem / Wo, x = rem - y * Wo;
  const half_t* p = in + ((size_t)img * C + c) * H * W + (2 * y) * W + 2 * x;
  float a = (float)p[0], b = (float)p[1], d0 = (float)p[W], e = (float)p[W + 1];
  out[((size_t)img * C + c) * Ho * Wo + y * Wo + x] =
      (half_t)fmaxf(fmaxf(a, b), fmaxf(d0, e));
}

// ----------------------------------------------- compressor FC: WMMA GEMM
// out[M=320][N=128] = leaky(A[M][K=18432](f16 row-major) * Bpack + bias).
// Grid (M/16, N/16), one wave per 16x16 tile. A-rows are streamed through
// LDS with async load-to-LDS (ASYNCcnt) in 16x64-half double-buffered
// stages; B fragments come from the packed (L2-resident) buffer.
__global__ __launch_bounds__(32) void gemm_wmma_kernel(
    const half_t* __restrict__ A, const half_t* __restrict__ Bpack,
    const float* __restrict__ bias, float* __restrict__ out,
    int N, int K, int nKC) {
  int mt = blockIdx.x, nt = blockIdx.y, lane = threadIdx.x;
  const int koff = (lane < 16) ? 0 : 8;
  const half_t* bpack = Bpack + ((size_t)nt * nKC * 32 + lane) * 16;

  v8f acc = {};
#if HAVE_ASYNC_LDS && defined(__gfx1250__)
  __shared__ __attribute__((aligned(32))) half_t Asb[2][16 * 64];
  const half_t* Abase = A + (size_t)(mt * 16) * K;
  // one pipeline stage = 16 rows x 64 halves = 128 b128 transfers (4/lane)
  auto issue = [&](int buf, int kbase) {
#pragma unroll
    for (int t = 0; t < 4; ++t) {
      int task = lane + t * 32;
      int row = task >> 3, seg = task & 7;
      const half_t* g = Abase + (size_t)row * K + kbase + seg * 8;
      half_t* l = &Asb[buf][row * 64 + seg * 8];
      __builtin_amdgcn_global_load_async_to_lds_b128(
          (__attribute__((address_space(1))) b128_t*)g,
          (__attribute__((address_space(3))) b128_t*)l, 0, 0);
    }
  };
  const int nStage = nKC >> 1;  // 64 halves (2 k-chunks) per stage
  issue(0, 0);
  for (int s = 0; s < nStage; ++s) {
    if (s + 1 < nStage) {
      issue((s + 1) & 1, (s + 1) * 64);
      __builtin_amdgcn_s_wait_asynccnt(4);  // stage s complete (in-order)
    } else {
      __builtin_amdgcn_s_wait_asynccnt(0);
    }
    const half_t* lbuf = Asb[s & 1];
#pragma unroll
    for (int kk = 0; kk < 2; ++kk) {
      const half_t* ap = lbuf + (lane & 15) * 64 + kk * 32 + koff;
      v16h a = cat8(*(const v8h*)ap, *(const v8h*)(ap + 16));
      v16h b = *(const v16h*)(bpack + (size_t)(s * 2 + kk) * 32 * 16);
      acc = __builtin_amdgcn_wmma_f32_16x16x32_f16(
          false, a, false, b, (short)0, acc, false, false);
    }
  }
#else
  // Fallback: fragments direct from global (L2).
  const half_t* arow = A + (size_t)(mt * 16 + (lane & 15)) * K;
  for (int kc = 0; kc < nKC; ++kc) {
    if (kc + 1 < nKC) __builtin_prefetch(arow + (size_t)(kc + 1) * 32, 0, 0);
    const half_t* ap = arow + kc * 32 + koff;
    v16h a = cat8(*(const v8h*)ap, *(const v8h*)(ap + 16));
    v16h b = *(const v16h*)(bpack + (size_t)kc * 32 * 16);
    acc = __builtin_amdgcn_wmma_f32_16x16x32_f16(
        false, a, false, b, (short)0, acc, false, false);
  }
#endif
  int col = nt * 16 + (lane & 15);
  int mbase = mt * 16 + ((lane < 16) ? 0 : 8);
#pragma unroll
  for (int r = 0; r < 8; ++r) {
    float o = acc[r] + bias[col];
    out[(size_t)(mbase + r) * N + col] = leakyf(o);
  }
}

// --------------------------------------------------- graph filter (tiny)
__global__ void feat_to_xg_kernel(const float* __restrict__ feat,
                                  float* __restrict__ xg) {
  // feat [B*20][128] -> xg [B][128][20]
  int i = blockIdx.x * blockDim.x + threadIdx.x;
  if (i >= 16 * 128 * 20) return;
  int b = i / 2560, rem = i - b * 2560;
  int f = rem / 20, n = rem - f * 20;
  xg[i] = feat[((size_t)b * 20 + n) * 128 + f];
}

__global__ void graph_shift_kernel(const float* __restrict__ xin,
                                   const float* __restrict__ S,
                                   float* __restrict__ xout) {
  // xout[b][g][n] = sum_m xin[b][g][m] * S[b][m][n]
  int i = blockIdx.x * blockDim.x + threadIdx.x;
  if (i >= 16 * 128 * 20) return;
  int b = i / 2560, rem = i - b * 2560;
  int g = rem / 20, n = rem - g * 20;
  const float* xi = xin + ((size_t)b * 128 + g) * 20;
  const float* Sb = S + (size_t)b * 400;
  float acc = 0.f;
#pragma unroll
  for (int m = 0; m < 20; ++m) acc += xi[m] * Sb[m * 20 + n];
  xout[i] = acc;
}

__global__ void graph_combine_kernel(const float* __restrict__ xg,
                                     const float* __restrict__ x1,
                                     const float* __restrict__ x2,
                                     const float* __restrict__ h,   // [F][3][F]
                                     const float* __restrict__ gb,
                                     const float* __restrict__ refs,
                                     const float* __restrict__ alphas,
                                     float* __restrict__ sf) {      // [B*N][148]
  int i = blockIdx.x * blockDim.x + threadIdx.x;
  if (i >= 16 * 20 * 128) return;
  int b = i / (20 * 128), rem = i - b * 20 * 128;
  int n = rem / 128, f = rem - n * 128;
  const float* hf = h + (size_t)f * 3 * 128;
  float acc = gb[f];
  for (int g = 0; g < 128; ++g) {
    size_t xi = ((size_t)b * 128 + g) * 20 + n;
    acc += hf[g] * xg[xi] + hf[128 + g] * x1[xi] + hf[256 + g] * x2[xi];
  }
  size_t rowbase = ((size_t)b * 20 + n) * 148;
  sf[rowbase + f] = leakyf(acc);
  if (f < 10) {  // repeated-cat padding: refs x10, alphas x10
    sf[rowbase + 128 + f] = refs[b * 20 + n];
    sf[rowbase + 138 + f] = alphas[b * 20 + n];
  }
}

// --------------------------------------------------------- action MLP
__global__ void mlp_kernel(const float* __restrict__ in,
                           const float* __restrict__ w,   // [din][dout]
                           const float* __restrict__ bias,
                           float* __restrict__ out,
                           int din, int dout, int rows, int do_leaky) {
  int i = blockIdx.x * blockDim.x + threadIdx.x;
  if (i >= rows * dout) return;
  int r = i / dout, j = i - r * dout;
  const float* x = in + (size_t)r * din;
  float acc = bias[j];
  for (int k = 0; k < din; ++k) acc += x[k] * w[(size_t)k * dout + j];
  out[i] = do_leaky ? leakyf(acc) : acc;
}

// ---------------------------------------------------------------------------
extern "C" void kernel_launch(void* const* d_in, const int* in_sizes, int n_in,
                              void* d_out, int out_size, void* d_ws, size_t ws_size,
                              hipStream_t stream) {
  (void)in_sizes; (void)n_in; (void)out_size; (void)ws_size;

  const float* x      = (const float*)d_in[0];
  const float* S      = (const float*)d_in[1];
  const float* refs   = (const float*)d_in[2];
  const float* alphas = (const float*)d_in[3];
  const float* conv_w[5], *conv_b[5], *bn_g[5], *bn_be[5], *bn_mu[5], *bn_var[5];
  for (int l = 0; l < 5; ++l) {
    conv_w[l] = (const float*)d_in[4 + l];
    conv_b[l] = (const float*)d_in[9 + l];
    bn_g[l]   = (const float*)d_in[14 + l];
    bn_be[l]  = (const float*)d_in[19 + l];
    bn_mu[l]  = (const float*)d_in[24 + l];
    bn_var[l] = (const float*)d_in[29 + l];
  }
  const float* comp_w = (const float*)d_in[34];
  const float* comp_b = (const float*)d_in[35];
  const float* gf_h   = (const float*)d_in[36];
  const float* gf_b   = (const float*)d_in[37];
  const float* act_w[3], *act_b[3];
  for (int l = 0; l < 3; ++l) {
    act_w[l] = (const float*)d_in[38 + l];
    act_b[l] = (const float*)d_in[41 + l];
  }

  // ---- workspace bump allocator (256B aligned) ----
  char* ws = (char*)d_ws;
  size_t off = 0;
  auto alloc = [&](size_t bytes) -> void* {
    off = (off + 255) & ~(size_t)255;
    void* p = ws + off;
    off += bytes;
    return p;
  };

  const int IMG = 320;
  const int CinT[5]  = {1, 32, 32, 64, 64};
  const int CoutT[5] = {32, 32, 64, 64, 128};

  half_t* bufA = (half_t*)alloc((size_t)51200000 * 2);   // max: conv2 out
  half_t* bufB = (half_t*)alloc((size_t)102400000 * 2);  // max: conv0 out
  half_t* wpack[5];
  float *scale[5], *shift[5];
  for (int l = 0; l < 5; ++l) {
    int Kfull = CinT[l] * 9, nKC = (Kfull + 31) / 32;
    wpack[l] = (half_t*)alloc((size_t)(CoutT[l] / 16) * nKC * 32 * 16 * 2);
    scale[l] = (float*)alloc(CoutT[l] * 4);
    shift[l] = (float*)alloc(CoutT[l] * 4);
  }
  half_t* compack = (half_t*)alloc((size_t)8 * 576 * 32 * 16 * 2);
  float* feat = (float*)alloc((size_t)320 * 128 * 4);
  float* xg   = (float*)alloc((size_t)40960 * 4);
  float* xk1  = (float*)alloc((size_t)40960 * 4);
  float* xk2  = (float*)alloc((size_t)40960 * 4);
  float* sf   = (float*)alloc((size_t)320 * 148 * 4);
  float* a1   = (float*)alloc((size_t)320 * 128 * 4);
  float* a2   = (float*)alloc((size_t)320 * 128 * 4);

  // ---- input f32 -> f16 ----
  {
    int n = IMG * 100 * 100;
    cvt_f32_f16_kernel<<<(n + 255) / 256, 256, 0, stream>>>(x, bufA, n);
  }

  // ---- 5 conv layers (+BN+Leaky fused), pool after 0,2,4 ----
  half_t* cur = bufA;
  half_t* nxt = bufB;
  int H = 100, W = 100;
  for (int l = 0; l < 5; ++l) {
    int Cin = CinT[l], Cout = CoutT[l];
    int Kfull = Cin * 9, nKC = (Kfull + 31) / 32;
    prep_bn_kernel<<<1, 128, 0, stream>>>(conv_b[l], bn_g[l], bn_be[l],
                                          bn_mu[l], bn_var[l],
                                          scale[l], shift[l], Cout);
    int rep = (Cout / 16) * nKC * 32;
    repack_conv_w_kernel<<<(rep + 255) / 256, 256, 0, stream>>>(
        conv_w[l], wpack[l], Cin, Cout, Kfull, nKC);
    int tiles = (H * W + CONV_NPIX - 1) / CONV_NPIX;
    conv_wmma_kernel<<<dim3(tiles, IMG), dim3(32, Cout / 16), 0, stream>>>(
        cur, wpack[l], scale[l], shift[l], nxt, Cin, Cout, H, W, Kfull, nKC);
    half_t* t = cur; cur = nxt; nxt = t;
    if ((l % 2) == 0) {
      int Ho = H / 2, Wo = W / 2;
      int tot = Cout * Ho * Wo;
      maxpool2_kernel<<<dim3((tot + 255) / 256, IMG), 256, 0, stream>>>(
          cur, nxt, Cout, H, W);
      t = cur; cur = nxt; nxt = t;
      H = Ho; W = Wo;
    }
  }
  // cur now holds pooled conv output: f16 [320][128*12*12] = [320][18432]

  // ---- compressor FC (WMMA GEMM, M=320 N=128 K=18432) ----
  {
    const int K = 18432, N = 128, nKC = K / 32;  // 576
    int rep = (N / 16) * nKC * 32;
    repack_gemm_b_kernel<<<(rep + 255) / 256, 256, 0, stream>>>(
        comp_w, compack, K, N, nKC);
    gemm_wmma_kernel<<<dim3(320 / 16, N / 16), 32, 0, stream>>>(
        cur, compack, comp_b, feat, N, K, nKC);
  }

  // ---- K=3 graph filter ----
  {
    int n = 16 * 128 * 20;
    feat_to_xg_kernel<<<(n + 255) / 256, 256, 0, stream>>>(feat, xg);
    graph_shift_kernel<<<(n + 255) / 256, 256, 0, stream>>>(xg, S, xk1);
    graph_shift_kernel<<<(n + 255) / 256, 256, 0, stream>>>(xk1, S, xk2);
    graph_combine_kernel<<<(16 * 20 * 128 + 255) / 256, 256, 0, stream>>>(
        xg, xk1, xk2, gf_h, gf_b, refs, alphas, sf);
  }

  // ---- action MLP 148 -> 128 -> 128 -> 2 ----
  {
    int n1 = 320 * 128;
    mlp_kernel<<<(n1 + 255) / 256, 256, 0, stream>>>(sf, act_w[0], act_b[0],
                                                     a1, 148, 128, 320, 1);
    mlp_kernel<<<(n1 + 255) / 256, 256, 0, stream>>>(a1, act_w[1], act_b[1],
                                                     a2, 128, 128, 320, 1);
    int n2 = 320 * 2;
    mlp_kernel<<<(n2 + 255) / 256, 256, 0, stream>>>(a2, act_w[2], act_b[2],
                                                     (float*)d_out, 128, 2, 320, 0);
  }
}
